// MultiHeadAttention_50964081934663
// MI455X (gfx1250) — matrix operand order repaired and checked by reading, not compiled, not run on hardware
//
#include <hip/hip_runtime.h>
#include <hip/hip_bf16.h>
#include <math.h>

// ---- problem constants (from reference) ----
#define BB 2
#define SS 2048
#define DD 1024
#define HH 16
#define HD 64
#define MT (BB * SS)          // 4096 rows in the "merged" GEMMs
#define SCALE 0.125f          // 1/sqrt(64)

typedef _Float16 half_t;
typedef __attribute__((ext_vector_type(16))) _Float16 v16h;
typedef __attribute__((ext_vector_type(8)))  _Float16 v8h;
typedef __attribute__((ext_vector_type(8)))  float    v8f;
typedef __attribute__((ext_vector_type(4)))  float    v4f;

// D = A(16x32 f16) * B(32x16 f16) + C(16x16 f32)
__device__ __forceinline__ v8f wmma_f16(v16h a, v16h b, v8f c) {
    return __builtin_amdgcn_wmma_f32_16x16x32_f16(
        /*neg_a=*/false, a, /*neg_b=*/false, b,
        /*c_mod=*/(short)0, c, /*reuse_a=*/false, /*reuse_b=*/false);
}

__device__ __forceinline__ void st_nt(float* p, float v) {
    __builtin_nontemporal_store(v, p);
}
__device__ __forceinline__ float ld_nt(const float* p) {
    return __builtin_nontemporal_load(p);
}

// ---------------- fragment loaders ----------------
// A fragment (16x32, f16 per-lane layout): lane holds row (lane&15).
// halves j = K (lane>>4)*16 + j
__device__ __forceinline__ v16h load_a(const half_t* __restrict__ p, int ld,
                                       int row0, int lane) {
    const int kb = ((lane >> 4) & 1) * 16;
    const half_t* q = p + (size_t)(row0 + (lane & 15)) * ld + kb;
    v8h lo = *(const v8h*)(q);
    v8h hi = *(const v8h*)(q + 8);
    v16h r;
#pragma unroll
    for (int i = 0; i < 8; ++i) { r[i] = lo[i]; r[i + 8] = hi[i]; }
    return r;
}

__device__ __forceinline__ v16h load_a(const float* __restrict__ p, int ld,
                                       int row0, int lane) {
    const int kb = ((lane >> 4) & 1) * 16;
    const float* q = p + (size_t)(row0 + (lane & 15)) * ld + kb;
    v4f x0 = *(const v4f*)(q),      x1 = *(const v4f*)(q + 4);
    v4f y0 = *(const v4f*)(q + 8), y1 = *(const v4f*)(q + 12);
    v16h r;
#pragma unroll
    for (int i = 0; i < 4; ++i) {
        r[i]      = (_Float16)x0[i];
        r[i + 4]  = (_Float16)x1[i];
        r[i + 8]  = (_Float16)y0[i];
        r[i + 12] = (_Float16)y1[i];
    }
    return r;
}

// B fragment (32x16): lane holds column (lane&15); halves j = K (lane>>4)*16 + j.
// Source matrix is stored row-major as Bsrc[n][k] (i.e. B[k][n] = Bsrc[n][k]),
// so the 16 halves are contiguous in memory.
__device__ __forceinline__ v16h load_b(const half_t* __restrict__ p, int ld,
                                       int col0, int lane) {
    const int kb = ((lane >> 4) & 1) * 16;
    const half_t* q = p + (size_t)(col0 + (lane & 15)) * ld + kb;
    v8h lo = *(const v8h*)(q);
    v8h hi = *(const v8h*)(q + 8);
    v16h r;
#pragma unroll
    for (int i = 0; i < 8; ++i) { r[i] = lo[i]; r[i + 8] = hi[i]; }
    return r;
}

__device__ __forceinline__ v16h load_b(const float* __restrict__ p, int ld,
                                       int col0, int lane) {
    const int kb = ((lane >> 4) & 1) * 16;
    const float* q = p + (size_t)(col0 + (lane & 15)) * ld + kb;
    v4f x0 = *(const v4f*)(q),     x1 = *(const v4f*)(q + 4);
    v4f x2 = *(const v4f*)(q + 8), x3 = *(const v4f*)(q + 12);
    v16h r;
#pragma unroll
    for (int i = 0; i < 4; ++i) {
        r[i]      = (_Float16)x0[i];
        r[i + 4]  = (_Float16)x1[i];
        r[i + 8]  = (_Float16)x2[i];
        r[i + 12] = (_Float16)x3[i];
    }
    return r;
}

// ---------------- GEMM: C[M,N] = X[M,K] * W[N,K]^T + bias ----------------
// MODE 0: out f16, split heads  [B,S,D] -> [B,H,S,HD]   (Q, K projections)
// MODE 1: out f16, transposed   [B,S,D] -> [B,H,HD,S]   (V projection)
// MODE 2: out f32, plain row-major [M,N]                 (output projection)
// 256 threads = 8 waves; block tile 128x128; wave tile 32x64 (2x4 WMMA frags:
// 8 WMMA per 6 fragment loads).
template <typename TIn, int MODE>
__global__ __launch_bounds__(256) void gemm_proj(
    const TIn* __restrict__ X, const float* __restrict__ W,
    const float* __restrict__ bias, void* __restrict__ out, int K) {
    const int lane = threadIdx.x & 31;
    const int wave = threadIdx.x >> 5;
    const int wm = wave & 3, wn = wave >> 2;
    const int m0 = blockIdx.x * 128 + wm * 32;
    const int n0 = blockIdx.y * 128 + wn * 64;

    v8f c[2][4] = {{{0}}};
    for (int k0 = 0; k0 < K; k0 += 32) {
        if (k0 + 32 < K) {  // prefetch next K-chunk (global_prefetch_b8)
            __builtin_prefetch(X + (size_t)(m0 + (lane & 15)) * K + k0 + 32, 0, 3);
            __builtin_prefetch(W + (size_t)(n0 + (lane & 15)) * K + k0 + 32, 0, 3);
        }
        v16h a0 = load_a(X + k0, K, m0,      lane);
        v16h a1 = load_a(X + k0, K, m0 + 16, lane);
#pragma unroll
        for (int j = 0; j < 4; ++j) {
            v16h bf = load_b(W + k0, K, n0 + j * 16, lane);
            c[0][j] = wmma_f16(a0, bf, c[0][j]);
            c[1][j] = wmma_f16(a1, bf, c[1][j]);
        }
    }

#pragma unroll
    for (int i = 0; i < 2; ++i)
#pragma unroll
        for (int j = 0; j < 4; ++j) {
            const int col = n0 + j * 16 + (lane & 15);
            const float bv = bias[col];
#pragma unroll
            for (int vr = 0; vr < 8; ++vr) {
                const int row = m0 + i * 16 + ((lane >> 4) & 1) * 8 + vr;
                const float val = c[i][j][vr] + bv;
                if (MODE == 2) {
                    ((float*)out)[(size_t)row * DD + col] = val;
                } else {
                    const int b = row >> 11, s = row & (SS - 1);
                    const int h = col >> 6,  hd = col & (HD - 1);
                    if (MODE == 0)
                        ((half_t*)out)[(((size_t)(b * HH + h)) * SS + s) * HD + hd] =
                            (half_t)val;
                    else
                        ((half_t*)out)[(((size_t)(b * HH + h)) * HD + hd) * SS + s] =
                            (half_t)val;
                }
            }
        }
}

// ---------------- scores = scale * Q K^T, masked -> attn (fp32) -----------
// grid: (B*H, S/128, S/64); 128 threads = 4 waves; each wave: 32x64 tile.
// attn is written with non-temporal stores (537 MB stream, > L2 capacity).
__global__ __launch_bounds__(128) void attn_scores(
    const half_t* __restrict__ Qh, const half_t* __restrict__ Kh,
    const int* __restrict__ mask, float* __restrict__ attn) {
    const int lane = threadIdx.x & 31;
    const int wave = threadIdx.x >> 5;
    const int bh = blockIdx.x;
    const int b = bh >> 4;
    const int q0 = blockIdx.y * 128 + wave * 32;
    const int n0 = blockIdx.z * 64;
    const half_t* Qb = Qh + (size_t)bh * SS * HD;
    const half_t* Kb = Kh + (size_t)bh * SS * HD;

    v8f c[2][4] = {{{0}}};
#pragma unroll
    for (int kc = 0; kc < 2; ++kc) {
        const int k0 = kc * 32;
        v16h a0 = load_a(Qb + k0, HD, q0,      lane);
        v16h a1 = load_a(Qb + k0, HD, q0 + 16, lane);
#pragma unroll
        for (int j = 0; j < 4; ++j) {
            v16h bf = load_b(Kb + k0, HD, n0 + j * 16, lane);
            c[0][j] = wmma_f16(a0, bf, c[0][j]);
            c[1][j] = wmma_f16(a1, bf, c[1][j]);
        }
    }

    float* ab = attn + (size_t)bh * SS * SS;
#pragma unroll
    for (int i = 0; i < 2; ++i)
#pragma unroll
        for (int j = 0; j < 4; ++j) {
            const int col = n0 + j * 16 + (lane & 15);
            const bool masked = (mask[b * SS + col] == 0);
#pragma unroll
            for (int vr = 0; vr < 8; ++vr) {
                const int row = q0 + i * 16 + ((lane >> 4) & 1) * 8 + vr;
                const float v = masked ? -1e9f : c[i][j][vr] * SCALE;
                st_nt(ab + (size_t)row * SS + col, v);
            }
        }
}

// ---------------- row softmax over attn (in place, NT streaming) ----------
__global__ __launch_bounds__(256) void softmax_rows(float* __restrict__ attn) {
    __shared__ float red[256];
    float* p = attn + (size_t)blockIdx.x * SS;
    const int t = threadIdx.x;

    float vals[8];
    float m = -INFINITY;
#pragma unroll
    for (int i = 0; i < 8; ++i) {
        vals[i] = ld_nt(p + t + i * 256);
        m = fmaxf(m, vals[i]);
    }
    red[t] = m;
    __syncthreads();
    for (int s = 128; s > 0; s >>= 1) {
        if (t < s) red[t] = fmaxf(red[t], red[t + s]);
        __syncthreads();
    }
    m = red[0];
    __syncthreads();

    float sum = 0.f;
#pragma unroll
    for (int i = 0; i < 8; ++i) {
        vals[i] = __expf(vals[i] - m);
        sum += vals[i];
    }
    red[t] = sum;
    __syncthreads();
    for (int s = 128; s > 0; s >>= 1) {
        if (t < s) red[t] += red[t + s];
        __syncthreads();
    }
    const float inv = 1.0f / red[0];
#pragma unroll
    for (int i = 0; i < 8; ++i) st_nt(p + t + i * 256, vals[i] * inv);
}

// ---------------- ctx = attn @ V  ([S,S] x [S,HD] per head) ---------------
// grid: (B*H, S/256); 256 threads = 8 waves; wave tile 32x64 over 256x64.
// VhT is [B,H,HD,S] so B-fragments are contiguous. ctx stored f16 [B,S,D].
__global__ __launch_bounds__(256) void attn_ctx(
    const float* __restrict__ attn, const half_t* __restrict__ VhT,
    half_t* __restrict__ ctx) {
    const int lane = threadIdx.x & 31;
    const int wave = threadIdx.x >> 5;
    const int bh = blockIdx.x;
    const int b = bh >> 4, h = bh & 15;
    const int m0 = blockIdx.y * 256 + wave * 32;  // q rows
    const float* Ab = attn + (size_t)bh * SS * SS;
    const half_t* Bb = VhT + (size_t)bh * HD * SS;

    v8f c[2][4] = {{{0}}};
    for (int k0 = 0; k0 < SS; k0 += 32) {
        if (k0 + 32 < SS) {
            __builtin_prefetch(Ab + (size_t)(m0 + (lane & 15)) * SS + k0 + 32, 0, 3);
            __builtin_prefetch(Bb + (size_t)(lane & 15) * SS + k0 + 32, 0, 3);
        }
        v16h a0 = load_a(Ab + k0, SS, m0,      lane);
        v16h a1 = load_a(Ab + k0, SS, m0 + 16, lane);
#pragma unroll
        for (int j = 0; j < 4; ++j) {
            v16h bf = load_b(Bb + k0, SS, j * 16, lane);
            c[0][j] = wmma_f16(a0, bf, c[0][j]);
            c[1][j] = wmma_f16(a1, bf, c[1][j]);
        }
    }

#pragma unroll
    for (int i = 0; i < 2; ++i)
#pragma unroll
        for (int j = 0; j < 4; ++j) {
            const int hd = j * 16 + (lane & 15);
#pragma unroll
            for (int vr = 0; vr < 8; ++vr) {
                const int q = m0 + i * 16 + ((lane >> 4) & 1) * 8 + vr;
                ctx[((size_t)b * SS + q) * DD + h * HD + hd] =
                    (half_t)c[i][j][vr];
            }
        }
}

// -------------------------- launch ---------------------------------------
extern "C" void kernel_launch(void* const* d_in, const int* in_sizes, int n_in,
                              void* d_out, int out_size, void* d_ws,
                              size_t ws_size, hipStream_t stream) {
    const float* q    = (const float*)d_in[0];
    const float* k    = (const float*)d_in[1];
    const float* v    = (const float*)d_in[2];
    const int*   mask = (const int*)d_in[3];
    const float* wq = (const float*)d_in[4];
    const float* bq = (const float*)d_in[5];
    const float* wk = (const float*)d_in[6];
    const float* bk = (const float*)d_in[7];
    const float* wv = (const float*)d_in[8];
    const float* bv = (const float*)d_in[9];
    const float* wo = (const float*)d_in[10];
    const float* bo = (const float*)d_in[11];

    float* out  = (float*)d_out;
    float* attn = out + (size_t)BB * SS * DD;   // second tuple element

    const size_t NE = (size_t)MT * DD;          // 4M elements per f16 buffer
    half_t* Qh  = (half_t*)d_ws;
    half_t* Kh  = Qh + NE;
    half_t* VhT = Kh + NE;
    half_t* ctx = VhT + NE;

    dim3 gP(MT / 128, DD / 128);                // 32 x 8
    gemm_proj<float, 0><<<gP, 256, 0, stream>>>(q, wq, bq, Qh,  DD);
    gemm_proj<float, 0><<<gP, 256, 0, stream>>>(k, wk, bk, Kh,  DD);
    gemm_proj<float, 1><<<gP, 256, 0, stream>>>(v, wv, bv, VhT, DD);

    attn_scores<<<dim3(BB * HH, SS / 128, SS / 64), 128, 0, stream>>>(
        Qh, Kh, mask, attn);

    softmax_rows<<<dim3(BB * HH * SS), 256, 0, stream>>>(attn);

    attn_ctx<<<dim3(BB * HH, SS / 256), 256, 0, stream>>>(attn, VhT, ctx);

    gemm_proj<half_t, 2><<<gP, 256, 0, stream>>>(ctx, wo, bo, out, DD);
}
